// EmformerEncoder_52209622450744
// MI455X (gfx1250) — matrix operand order repaired and checked by reading, not compiled, not run on hardware
//
#include <hip/hip_runtime.h>

typedef _Float16 h16;
typedef __attribute__((ext_vector_type(8)))  _Float16 v8h;
typedef __attribute__((ext_vector_type(16))) _Float16 v16h;
typedef __attribute__((ext_vector_type(8)))  float    v8f;

#define D_    512
#define H_    8
#define HD_   64
#define S_    32
#define B_    32
#define F_    2048
#define NL_   8
#define NSEG_ 16
#define LCAP_ 8
#define BD_   (B_*D_)        /* 16384 */
#define SEGE_ (S_*B_*D_)     /* 524288 */
#define TKP_  320
#define SHLD_ 328

// ---------------- fragment loaders (CDNA5 WMMA VGPR layouts) ----------------
// A 16x32 f16: lane lh=lane&15 -> M=lh ; hi=lane>>4 selects K halves
//   elems 0..7  = K[hi*8 .. hi*8+7], elems 8..15 = K[16+hi*8 .. 16+hi*8+7]
__device__ inline v16h frag_a(const h16* rowbase, int ld, int lane) {
  int lh = lane & 15, hi = lane >> 4;
  const h16* p = rowbase + lh * ld + hi * 8;
  v8h x0 = *(const v8h*)p;
  v8h x1 = *(const v8h*)(p + 16);
  v16h r;
#pragma unroll
  for (int i = 0; i < 8; ++i) { r[i] = x0[i]; r[i + 8] = x1[i]; }
  return r;
}
// B 32x16 f16 staged transposed (row = N, 32 contiguous K per row):
//   lane lh -> N=lh ; elems 0..7 = K[hi*8 .. hi*8+7], elems 8..15 = K[16+hi*8 .. 16+hi*8+7]
__device__ inline v16h frag_b(const h16* rowbase, int ld, int lane) {
  int lh = lane & 15, hi = lane >> 4;
  const h16* p = rowbase + lh * ld + hi * 8;
  v8h x0 = *(const v8h*)p;
  v8h x1 = *(const v8h*)(p + 16);
  v16h r;
#pragma unroll
  for (int i = 0; i < 8; ++i) { r[i] = x0[i]; r[i + 8] = x1[i]; }
  return r;
}

// ---------------- generic GEMM: C[M,N] = A[M,K] * W[N,K]^T + bias (+resid)(relu) ----------------
// BM=64 x BN=128, K staged 64/iter, ping-pong LDS. Activation tile: load f32 ->
// v_cvt -> ds_store. Weight tile: global_load_async_to_lds_b128 (ASYNCcnt),
// synchronized with s_wait_asynccnt before the block barrier.
__global__ __launch_bounds__(256) void gemm_wmma(
    const float* __restrict__ A, const h16* __restrict__ W,
    const float* __restrict__ bias, const float* __restrict__ resid,
    float* __restrict__ C, int M, int N, int K, int relu)
{
  __shared__ h16 As[2][64 * 72];
  __shared__ h16 Ws[2][128 * 72];
  const int tid  = threadIdx.x;
  const int lane = tid & 31;
  const int wave = tid >> 5;
  const int m0 = blockIdx.y * 64;
  const int n0 = blockIdx.x * 128;
  const int r  = (wave >> 2) * 32;   // row offset of this wave's 32x32 block
  const int cq = (wave & 3) * 32;    // col offset

  v8f acc00 = {}, acc01 = {}, acc10 = {}, acc11 = {};

  auto load_stage = [&](int buf, int k0) {
    { // A tile 64x64: f32 -> f16 -> LDS
      int row = tid >> 2;
      int c   = (tid & 3) * 16;
      int gm  = m0 + row;
      h16* dst = &As[buf][row * 72 + c];
      if (gm < M) {
        const float4* src = (const float4*)(A + (size_t)gm * K + k0 + c);
        float4 f0 = src[0], f1 = src[1], f2 = src[2], f3 = src[3];
        v8h h0, h1;
        h0[0]=(h16)f0.x; h0[1]=(h16)f0.y; h0[2]=(h16)f0.z; h0[3]=(h16)f0.w;
        h0[4]=(h16)f1.x; h0[5]=(h16)f1.y; h0[6]=(h16)f1.z; h0[7]=(h16)f1.w;
        h1[0]=(h16)f2.x; h1[1]=(h16)f2.y; h1[2]=(h16)f2.z; h1[3]=(h16)f2.w;
        h1[4]=(h16)f3.x; h1[5]=(h16)f3.y; h1[6]=(h16)f3.z; h1[7]=(h16)f3.w;
        *(v8h*)dst       = h0;
        *(v8h*)(dst + 8) = h1;
      } else {
        v8h z = {};
        *(v8h*)dst       = z;
        *(v8h*)(dst + 8) = z;
      }
    }
    { // W tile 128x64: async DMA global -> LDS, 64B per lane
      int row = tid >> 1;
      int c32 = (tid & 1) * 32;
      const h16* src = W + (size_t)(n0 + row) * K + k0 + c32;
      unsigned lds = (unsigned)(uintptr_t)(&Ws[buf][row * 72 + c32]);
      asm volatile(
          "global_load_async_to_lds_b128 %0, %1, off\n\t"
          "global_load_async_to_lds_b128 %0, %1, off offset:16\n\t"
          "global_load_async_to_lds_b128 %0, %1, off offset:32\n\t"
          "global_load_async_to_lds_b128 %0, %1, off offset:48"
          :: "v"(lds), "v"(src) : "memory");
    }
  };

  load_stage(0, 0);
  int cur = 0;
  for (int k0 = 0; k0 < K; k0 += 64) {
    asm volatile("s_wait_asynccnt 0x0" ::: "memory");  // wave's async W tile done
    __syncthreads();                                    // everyone's tile visible
    if (k0 + 64 < K) load_stage(cur ^ 1, k0 + 64);      // prefetch next stage
#pragma unroll
    for (int kk = 0; kk < 64; kk += 32) {
      v16h a0 = frag_a(&As[cur][(r +  0) * 72 + kk], 72, lane);
      v16h a1 = frag_a(&As[cur][(r + 16) * 72 + kk], 72, lane);
      v16h b0 = frag_b(&Ws[cur][(cq +  0) * 72 + kk], 72, lane);
      v16h b1 = frag_b(&Ws[cur][(cq + 16) * 72 + kk], 72, lane);
      acc00 = __builtin_amdgcn_wmma_f32_16x16x32_f16(false, a0, false, b0, (short)0, acc00, false, false);
      acc01 = __builtin_amdgcn_wmma_f32_16x16x32_f16(false, a0, false, b1, (short)0, acc01, false, false);
      acc10 = __builtin_amdgcn_wmma_f32_16x16x32_f16(false, a1, false, b0, (short)0, acc10, false, false);
      acc11 = __builtin_amdgcn_wmma_f32_16x16x32_f16(false, a1, false, b1, (short)0, acc11, false, false);
    }
    cur ^= 1;
  }

  const int lh = lane & 15, hi = lane >> 4;
  v8f* accs[4] = { &acc00, &acc01, &acc10, &acc11 };
#pragma unroll
  for (int t = 0; t < 4; ++t) {
    int ti = t >> 1, tj = t & 1;
    int col = n0 + cq + tj * 16 + lh;
    float bv = bias ? bias[col] : 0.f;
#pragma unroll
    for (int i = 0; i < 8; ++i) {
      int row = m0 + r + ti * 16 + hi * 8 + i;
      if (row < M) {
        float v = (*accs[t])[i] + bv;
        if (resid) v += resid[(size_t)row * N + col];
        if (relu)  v = v > 0.f ? v : 0.f;
        C[(size_t)row * N + col] = v;
      }
    }
  }
}

// ---------------- fused attention per (b,h): softmax(QK^T/8) V ----------------
__global__ __launch_bounds__(256) void attn_wmma(
    const float* __restrict__ qp, const float* __restrict__ kp,
    const float* __restrict__ vp, float* __restrict__ o, int Tk)
{
  __shared__ h16 Qs[32 * 72];
  __shared__ h16 Ks[64 * 72];          // reused for V^T chunks
  __shared__ h16 Sh[32 * SHLD_];
  const int tid  = threadIdx.x;
  const int lane = tid & 31;
  const int wave = tid >> 5;
  const int b = blockIdx.x >> 3;
  const int h = blockIdx.x & 7;
  const int nchunk = (Tk + 63) >> 6;
  const int qt   = wave >> 2;          // q tile 0..1
  const int wsub = wave & 3;           // token/d tile 0..3

  { // stage Q 32x64
    int q  = tid >> 3;
    int d0 = (tid & 7) * 8;
    const float* src = qp + ((size_t)q * B_ + b) * D_ + h * HD_ + d0;
    h16* dst = &Qs[q * 72 + d0];
#pragma unroll
    for (int i = 0; i < 8; ++i) dst[i] = (h16)src[i];
  }
  __syncthreads();

  // ---- scores S = QK^T * 0.125, written f16 into Sh ----
  for (int cc = 0; cc < nchunk; ++cc) {
    { // stage K chunk: 64 tokens x 64 dims, row = token
      int tk = tid >> 2;
      int d0 = (tid & 3) * 16;
      int gtk = cc * 64 + tk;
      h16* dst = &Ks[tk * 72 + d0];
      if (gtk < Tk) {
        const float* src = kp + ((size_t)gtk * B_ + b) * D_ + h * HD_ + d0;
#pragma unroll
        for (int i = 0; i < 16; ++i) dst[i] = (h16)src[i];
      } else {
#pragma unroll
        for (int i = 0; i < 16; ++i) dst[i] = (h16)0.f;
      }
    }
    __syncthreads();
    v8f sacc = {};
#pragma unroll
    for (int kc = 0; kc < 2; ++kc) {
      v16h aq = frag_a(&Qs[(qt * 16) * 72 + kc * 32], 72, lane);
      v16h bk = frag_b(&Ks[(wsub * 16) * 72 + kc * 32], 72, lane);
      sacc = __builtin_amdgcn_wmma_f32_16x16x32_f16(false, aq, false, bk, (short)0, sacc, false, false);
    }
    {
      int lh = lane & 15, hi = lane >> 4;
      int tok = cc * 64 + wsub * 16 + lh;
#pragma unroll
      for (int i = 0; i < 8; ++i) {
        int q = qt * 16 + hi * 8 + i;
        Sh[q * SHLD_ + tok] = (h16)(sacc[i] * 0.125f);
      }
    }
    __syncthreads();
  }

  // ---- softmax over valid tokens, zero the pad ----
  if (tid < 32) {
    h16* rowp = &Sh[tid * SHLD_];
    float mx = -1e30f;
    for (int k = 0; k < Tk; ++k) { float v = (float)rowp[k]; mx = v > mx ? v : mx; }
    float sum = 0.f;
    for (int k = 0; k < Tk; ++k) { float e = __expf((float)rowp[k] - mx); sum += e; rowp[k] = (h16)e; }
    float inv = 1.f / sum;
    for (int k = 0; k < Tk; ++k) rowp[k] = (h16)((float)rowp[k] * inv);
    for (int k = Tk; k < TKP_; ++k) rowp[k] = (h16)0.f;
  }
  __syncthreads();

  // ---- O = S V ; each wave owns one 16(q) x 16(d) tile ----
  v8f oacc = {};
  for (int cc = 0; cc < nchunk; ++cc) {
    { // stage V^T chunk: Vt[d][tok]
      int tk = tid >> 2;
      int d0 = (tid & 3) * 16;
      int gtk = cc * 64 + tk;
      if (gtk < Tk) {
        const float* src = vp + ((size_t)gtk * B_ + b) * D_ + h * HD_ + d0;
#pragma unroll
        for (int i = 0; i < 16; ++i) Ks[(d0 + i) * 72 + tk] = (h16)src[i];
      } else {
#pragma unroll
        for (int i = 0; i < 16; ++i) Ks[(d0 + i) * 72 + tk] = (h16)0.f;
      }
    }
    __syncthreads();
#pragma unroll
    for (int sub = 0; sub < 2; ++sub) {
      v16h as = frag_a(&Sh[(qt * 16) * SHLD_ + cc * 64 + sub * 32], SHLD_, lane);
      v16h bv = frag_b(&Ks[(wsub * 16) * 72 + sub * 32], 72, lane);
      oacc = __builtin_amdgcn_wmma_f32_16x16x32_f16(false, as, false, bv, (short)0, oacc, false, false);
    }
    __syncthreads();
  }
  {
    int lh = lane & 15, hi = lane >> 4;
    int d = wsub * 16 + lh;
#pragma unroll
    for (int i = 0; i < 8; ++i) {
      int q = qt * 16 + hi * 8 + i;
      o[((size_t)q * B_ + b) * D_ + h * HD_ + d] = oacc[i];
    }
  }
}

// ---------------- LayerNorm over D=512, one row per block ----------------
__global__ __launch_bounds__(256) void ln_kernel(
    const float* __restrict__ x, const float* __restrict__ g,
    const float* __restrict__ bt, float* __restrict__ y)
{
  __shared__ float red[256];
  const int row = blockIdx.x;
  const float* xr = x + (size_t)row * D_;
  float v0 = xr[threadIdx.x], v1 = xr[threadIdx.x + 256];
  red[threadIdx.x] = v0 + v1;
  __syncthreads();
  for (int s = 128; s > 0; s >>= 1) { if (threadIdx.x < s) red[threadIdx.x] += red[threadIdx.x + s]; __syncthreads(); }
  float mean = red[0] * (1.f / 512.f);
  __syncthreads();
  float d0 = v0 - mean, d1 = v1 - mean;
  red[threadIdx.x] = d0 * d0 + d1 * d1;
  __syncthreads();
  for (int s = 128; s > 0; s >>= 1) { if (threadIdx.x < s) red[threadIdx.x] += red[threadIdx.x + s]; __syncthreads(); }
  float inv = rsqrtf(red[0] * (1.f / 512.f) + 1e-5f);
  float* yr = y + (size_t)row * D_;
  yr[threadIdx.x]       = d0 * inv * g[threadIdx.x]       + bt[threadIdx.x];
  yr[threadIdx.x + 256] = d1 * inv * g[threadIdx.x + 256] + bt[threadIdx.x + 256];
}

// ---------------- kv assembly: [memory?][left ring][center] -> f32 [Tk,B,D] ----------------
__global__ void build_kv_kernel(float* __restrict__ kv, const float* __restrict__ mem,
                                const h16* __restrict__ ringL, const float* __restrict__ center,
                                int Tk, int L, int s, int hasMem)
{
  int idx = blockIdx.x * 256 + threadIdx.x;
  int total = Tk * BD_;
  if (idx >= total) return;
  int tk = idx / BD_;
  int rr = idx - tk * BD_;
  float v;
  if (hasMem && tk == 0) {
    v = mem[rr];
  } else {
    int j = tk - hasMem;
    if (j < L * S_) {
      int seg  = s - L + (j >> 5);
      int slot = seg & 7;
      int t    = j & 31;
      v = (float)ringL[(size_t)slot * SEGE_ + (size_t)t * BD_ + rr];
    } else {
      int t = j - L * S_;
      v = center[(size_t)t * BD_ + rr];
    }
  }
  kv[idx] = v;
}

__global__ void kv_mean_kernel(const float* __restrict__ kv, float* __restrict__ out, int Tk)
{
  int idx = blockIdx.x * 256 + threadIdx.x;   // b*D + d
  if (idx >= BD_) return;
  float s = 0.f;
  for (int t = 0; t < Tk; ++t) s += kv[(size_t)t * BD_ + idx];
  out[idx] = s / (float)Tk;
}

__global__ void f32_to_f16_kernel(const float* __restrict__ in, h16* __restrict__ out, int n)
{
  int idx = blockIdx.x * 256 + threadIdx.x;
  if (idx < n) out[idx] = (h16)in[idx];
}

// ---------------- host orchestration ----------------
extern "C" void kernel_launch(void* const* d_in, const int* in_sizes, int n_in,
                              void* d_out, int out_size, void* d_ws, size_t ws_size,
                              hipStream_t stream)
{
  (void)in_sizes; (void)n_in; (void)out_size; (void)ws_size;
  const float* x   = (const float*)d_in[0];
  const float* ipw = (const float*)d_in[1];
  const float* ipb = (const float*)d_in[2];
  const float* ow  = (const float*)d_in[3];
  const float* obv = (const float*)d_in[4];
  const float* g1  = (const float*)d_in[5];
  const float* b1  = (const float*)d_in[6];
  const float* g2  = (const float*)d_in[7];
  const float* b2  = (const float*)d_in[8];
  const float* w1  = (const float*)d_in[9];
  const float* bf1 = (const float*)d_in[10];
  const float* w2  = (const float*)d_in[11];
  const float* bf2 = (const float*)d_in[12];
  const float* mw  = (const float*)d_in[13];
  const float* mb  = (const float*)d_in[14];
  float* out = (float*)d_out;

  char* p = (char*)d_ws;
  auto carve = [&](size_t bytes) -> char* {
    char* r = p; p += (bytes + 255) & ~(size_t)255; return r;
  };
  const size_t ipw_n = (size_t)NL_ * 3 * D_ * D_;
  const size_t ow_n  = (size_t)NL_ * D_ * D_;
  const size_t w1_n  = (size_t)NL_ * F_ * D_;
  const size_t w2_n  = (size_t)NL_ * D_ * F_;
  const size_t mw_n  = (size_t)NL_ * D_ * D_;

  h16*   ipw16 = (h16*)carve(ipw_n * 2);
  h16*   ow16  = (h16*)carve(ow_n * 2);
  h16*   w116  = (h16*)carve(w1_n * 2);
  h16*   w216  = (h16*)carve(w2_n * 2);
  h16*   mw16  = (h16*)carve(mw_n * 2);
  h16*   ring  = (h16*)carve((size_t)NL_ * 8 * SEGE_ * 2);
  float* mems  = (float*)carve((size_t)NL_ * BD_ * 4);
  float* cbuf  = (float*)carve((size_t)SEGE_ * 4);
  float* qln   = (float*)carve((size_t)SEGE_ * 4);
  float* qpb   = (float*)carve((size_t)SEGE_ * 4);
  float* kvb   = (float*)carve((size_t)304 * BD_ * 4);
  float* kpb   = (float*)carve((size_t)304 * BD_ * 4);
  float* vpb   = (float*)carve((size_t)304 * BD_ * 4);
  float* kvm   = (float*)carve((size_t)BD_ * 4);
  float* attno = (float*)carve((size_t)SEGE_ * 4);
  float* x1b   = (float*)carve((size_t)SEGE_ * 4);
  float* ylb   = (float*)carve((size_t)SEGE_ * 4);
  float* hb    = (float*)carve((size_t)S_ * B_ * F_ * 4);

  auto cvt = [&](const float* src, h16* dst, size_t n) {
    f32_to_f16_kernel<<<(unsigned)((n + 255) / 256), 256, 0, stream>>>(src, dst, (int)n);
  };
  cvt(ipw, ipw16, ipw_n);
  cvt(ow,  ow16,  ow_n);
  cvt(w1,  w116,  w1_n);
  cvt(w2,  w216,  w2_n);
  cvt(mw,  mw16,  mw_n);

  auto gemm = [&](const float* A, const h16* W, const float* bias, const float* resid,
                  float* C, int M, int N, int K, int relu) {
    dim3 grid((unsigned)(N / 128), (unsigned)((M + 63) / 64));
    gemm_wmma<<<grid, 256, 0, stream>>>(A, W, bias, resid, C, M, N, K, relu);
  };

  const int Mq = S_ * B_;   // 1024
  for (int s = 0; s < NSEG_; ++s) {
    for (int li = 0; li < NL_; ++li) {
      const float* center = (li == 0) ? (x + (size_t)s * SEGE_) : cbuf;
      float* outptr = (li == NL_ - 1) ? (out + (size_t)s * SEGE_) : cbuf;
      const int L = s < LCAP_ ? s : LCAP_;
      const int hasMem = (li > 0) ? 1 : 0;
      const int Tk = hasMem + L * S_ + S_;
      h16* ringL = ring + (size_t)li * 8 * SEGE_;

      // 1. assemble kv = [memory, left, center]   (must precede ring push)
      build_kv_kernel<<<(unsigned)((Tk * BD_ + 255) / 256), 256, 0, stream>>>(
          kvb, hasMem ? (mems + (size_t)(li - 1) * BD_) : nullptr,
          ringL, center, Tk, L, s, hasMem);
      // 2. push current center into ring cache (f16)
      cvt(center, ringL + (size_t)(s & 7) * SEGE_, SEGE_);
      // 3. q = LN1(center)
      ln_kernel<<<Mq, 256, 0, stream>>>(center, g1 + li * D_, b1 + li * D_, qln);
      // 4-5. projections
      gemm(qln, ipw16 + (size_t)li * 3 * D_ * D_,                     ipb + (size_t)li * 3 * D_,          nullptr, qpb, Mq,       D_, D_, 0);
      gemm(kvb, ipw16 + (size_t)li * 3 * D_ * D_ + (size_t)D_ * D_,   ipb + (size_t)li * 3 * D_ + D_,     nullptr, kpb, Tk * B_,  D_, D_, 0);
      gemm(kvb, ipw16 + (size_t)li * 3 * D_ * D_ + (size_t)2*D_*D_,   ipb + (size_t)li * 3 * D_ + 2*D_,   nullptr, vpb, Tk * B_,  D_, D_, 0);
      // 6. memory output = mean(kv) @ mw^T + mb
      kv_mean_kernel<<<(unsigned)((BD_ + 255) / 256), 256, 0, stream>>>(kvb, kvm, Tk);
      gemm(kvm, mw16 + (size_t)li * D_ * D_, mb + (size_t)li * D_, nullptr, mems + (size_t)li * BD_, B_, D_, D_, 0);
      // 7. attention
      attn_wmma<<<B_ * H_, 256, 0, stream>>>(qpb, kpb, vpb, attno, Tk);
      // 8. x1 = center + o @ ow^T + ob
      gemm(attno, ow16 + (size_t)li * D_ * D_, obv + (size_t)li * D_, center, x1b, Mq, D_, D_, 0);
      // 9. y = LN2(x1)
      ln_kernel<<<Mq, 256, 0, stream>>>(x1b, g2 + li * D_, b2 + li * D_, ylb);
      // 10. h = relu(y @ w1^T + bf1)
      gemm(ylb, w116 + (size_t)li * F_ * D_, bf1 + (size_t)li * F_, nullptr, hb, Mq, F_, D_, 1);
      // 11. out = x1 + h @ w2^T + bf2
      gemm(hb, w216 + (size_t)li * D_ * F_, bf2 + (size_t)li * D_, x1b, outptr, Mq, D_, F_, 0);
    }
  }
}
